// MultiHeadSelfAttention_25168508354971
// MI455X (gfx1250) — matrix operand order repaired and checked by reading, not compiled, not run on hardware
//
#include <hip/hip_runtime.h>
#include <hip/hip_bf16.h>

// MI455X / gfx1250: wave32, WMMA 16x16x32 bf16 -> f32.
typedef __attribute__((ext_vector_type(16))) __bf16 v16bf;
typedef __attribute__((ext_vector_type(8)))  float  v8f;
typedef __attribute__((ext_vector_type(4)))  int    v4i;

#define AS_GLOBAL __attribute__((address_space(1)))
#define AS_LOCAL  __attribute__((address_space(3)))

#if defined(__HIP_DEVICE_COMPILE__) && defined(__has_builtin)
#  if __has_builtin(__builtin_amdgcn_global_load_async_to_lds_b128)
#    define HAVE_ASYNC_LDS 1
#  endif
#endif
#ifndef HAVE_ASYNC_LDS
#  define HAVE_ASYNC_LDS 0
#endif

__device__ __forceinline__ void async_wait_all() {
#if HAVE_ASYNC_LDS
#  if __has_builtin(__builtin_amdgcn_s_wait_asynccnt)
  __builtin_amdgcn_s_wait_asynccnt(0);
#  else
  asm volatile("s_wait_asynccnt 0" ::: "memory");
#  endif
#endif
}

__device__ __forceinline__ unsigned short f32_to_bf16_raw(float f) {
  unsigned int u = __builtin_bit_cast(unsigned int, f);
  if ((u & 0x7F800000u) != 0x7F800000u)        // finite: round-to-nearest-even
    u += 0x7FFFu + ((u >> 16) & 1u);
  return (unsigned short)(u >> 16);
}
__device__ __forceinline__ __bf16 bf16_from_raw(unsigned short r) {
  return __builtin_bit_cast(__bf16, r);
}
__device__ __forceinline__ v8f wmma_bf16(v16bf a, v16bf b, v8f c) {
  return __builtin_amdgcn_wmma_f32_16x16x32_bf16(false, a, false, b, (short)0, c,
                                                 false, false);
}

// ---------------------------------------------------------------------------
// 128x128x32-tiled bf16 WMMA GEMM, 256 threads = 8 waves (4x2 wave grid),
// each wave: 32x64 subtile = 2x4 WMMA accumulators. Software-pipelined:
// next K-tile is fetched to registers while the current LDS tile is consumed.
// B tile is stored K-contiguous (transposed) in LDS so B-fragment builds are
// ds_load_b128 runs instead of scalar u16 column walks.
// MODE 0: A = fp32 x [M,K]; B = fp32 W_qkv [K,N]; epilogue: +bias, scatter bf16
//         q [B,H,T,D] (pre-scaled 1/8), k [B,H,T,D], v TRANSPOSED [B,H,D,T].
// MODE 1: A = bf16 attn [M,K]; B = fp32 W_out [K,N]; epilogue: +bias, f32 out.
// ---------------------------------------------------------------------------
template <int MODE>
__global__ __launch_bounds__(256) void gemm_wmma_bf16(
    const void* __restrict__ Ap, const float* __restrict__ Bw,
    const float* __restrict__ bias,
    unsigned short* __restrict__ qd, unsigned short* __restrict__ kd,
    unsigned short* __restrict__ vd, float* __restrict__ fd,
    int M, int N, int K) {
  constexpr int BM = 128, BN = 128, BK = 32;
  __shared__ __align__(16) unsigned short As[BM][BK + 8];   // 80B rows: aligned,
  __shared__ __align__(16) unsigned short BsT[BN][BK + 8];  // conflict-spread

  const int tid = threadIdx.x;
  const int wv = tid >> 5, lane = tid & 31;
  const int hi = lane >> 4, l16 = lane & 15;
  const int wm = wv & 3, wn = wv >> 2;           // 4x2 wave grid
  const int m0 = blockIdx.x * BM, n0 = blockIdx.y * BN;

  const float*          Af = (const float*)Ap;
  const unsigned short* Ah = (const unsigned short*)Ap;

  float4  arF[4];
  ushort4 arH[4];
  float4  brF[4];

  auto gload = [&](int k0) {
#pragma unroll
    for (int it = 0; it < 4; ++it) {
      const int idx = it * 256 + tid;
      const int arow = idx >> 3, akc = (idx & 7) << 2;          // A: 8 thr/row
      if (MODE == 0)
        arF[it] = *(const float4*)(Af + (size_t)(m0 + arow) * K + (k0 + akc));
      else
        arH[it] = *(const ushort4*)(Ah + (size_t)(m0 + arow) * K + (k0 + akc));
      const int brow = idx >> 5, bnc = (idx & 31) << 2;         // B: 32 thr/row
      brF[it] = *(const float4*)(Bw + (size_t)(k0 + brow) * N + (n0 + bnc));
    }
  };
  auto lstore = [&]() {
#pragma unroll
    for (int it = 0; it < 4; ++it) {
      const int idx = it * 256 + tid;
      const int arow = idx >> 3, akc = (idx & 7) << 2;
      if (MODE == 0) {
        As[arow][akc + 0] = f32_to_bf16_raw(arF[it].x);
        As[arow][akc + 1] = f32_to_bf16_raw(arF[it].y);
        As[arow][akc + 2] = f32_to_bf16_raw(arF[it].z);
        As[arow][akc + 3] = f32_to_bf16_raw(arF[it].w);
      } else {
        As[arow][akc + 0] = arH[it].x; As[arow][akc + 1] = arH[it].y;
        As[arow][akc + 2] = arH[it].z; As[arow][akc + 3] = arH[it].w;
      }
      const int brow = idx >> 5, bnc = (idx & 31) << 2;
      BsT[bnc + 0][brow] = f32_to_bf16_raw(brF[it].x);          // transposed store
      BsT[bnc + 1][brow] = f32_to_bf16_raw(brF[it].y);
      BsT[bnc + 2][brow] = f32_to_bf16_raw(brF[it].z);
      BsT[bnc + 3][brow] = f32_to_bf16_raw(brF[it].w);
    }
  };

  v8f acc[2][4] = {};

  gload(0);
  lstore();
  for (int k0 = 0; k0 < K; k0 += BK) {
    const bool has_next = (k0 + BK) < K;
    if (has_next) gload(k0 + BK);      // overlap global latency with WMMAs
    __syncthreads();

    v16bf a[2], b[4];
#pragma unroll
    for (int i = 0; i < 2; ++i) {
      const int row = wm * 32 + i * 16 + l16;                   // A: lane%16 = M row
#pragma unroll
      for (int e = 0; e < 16; ++e) {
        const int kl = e + hi * 16;                             // contiguous in k
        a[i][e] = bf16_from_raw(As[row][kl]);
      }
    }
#pragma unroll
    for (int jb = 0; jb < 4; ++jb) {
      const int col = wn * 64 + jb * 16 + l16;                  // B: lane%16 = N col
#pragma unroll
      for (int e = 0; e < 16; ++e)
        b[jb][e] = bf16_from_raw(BsT[col][e + hi * 16]);        // contiguous in k
    }
#pragma unroll
    for (int i = 0; i < 2; ++i)
#pragma unroll
      for (int jb = 0; jb < 4; ++jb)
        acc[i][jb] = wmma_bf16(a[i], b[jb], acc[i][jb]);

    __syncthreads();
    if (has_next) lstore();
  }

  // Epilogue. C/D layout: M = r + 8*(lane/16), N = lane%16.
#pragma unroll
  for (int jb = 0; jb < 4; ++jb) {
    const int gn = n0 + wn * 64 + jb * 16 + l16;
    const float bv = bias[gn];
#pragma unroll
    for (int i = 0; i < 2; ++i) {
#pragma unroll
      for (int r = 0; r < 8; ++r) {
        const int gm = m0 + wm * 32 + i * 16 + hi * 8 + r;
        float val = acc[i][jb][r] + bv;
        if (MODE == 0) {
          const int which = gn >> 10, cc = gn & 1023;           // which of q/k/v
          const int h = cc >> 6, d = cc & 63;
          const int bb = gm >> 11, t = gm & 2047;               // T = 2048
          if (which == 0) {                                     // q: [B,H,T,D], *1/8
            val *= 0.125f;
            qd[(((size_t)(bb * 16 + h)) * 2048 + t) * 64 + d] = f32_to_bf16_raw(val);
          } else if (which == 1) {                              // k: [B,H,T,D]
            kd[(((size_t)(bb * 16 + h)) * 2048 + t) * 64 + d] = f32_to_bf16_raw(val);
          } else {                                              // v: [B,H,D,T]
            vd[(((size_t)(bb * 16 + h)) * 64 + d) * 2048 + t] = f32_to_bf16_raw(val);
          }
        } else {
          fd[(size_t)gm * N + gn] = val;
        }
      }
    }
  }
}

// ---------------------------------------------------------------------------
// Flash attention forward. Grid = (T/128, B*H), block = 256 (8 waves).
// Each wave owns 16 query rows; 128-key K / V^T tiles stream through double-
// buffered LDS via async-to-LDS DMA (ASYNCcnt), overlapping tile j+1's copy
// with tile j's 32 WMMAs + softmax. V is pre-transposed in global ([B,H,D,T])
// so both K and V^T fragment builds are contiguous ds_load_b128 runs.
// ---------------------------------------------------------------------------
__global__ __launch_bounds__(256) void attn_wmma(
    const unsigned short* __restrict__ Q, const unsigned short* __restrict__ Kg,
    const unsigned short* __restrict__ Vt, unsigned short* __restrict__ Og) {
  constexpr int T = 2048, D = 64, C = 1024;
  constexpr int KP = D + 8;                      // 144B rows (9x16): aligned
  constexpr int VP = 128 + 8;                    // 272B rows (17x16): aligned
  __shared__ __align__(16) unsigned short Ks[2][128][KP];   // 36 KB
  __shared__ __align__(16) unsigned short VsT[2][64][VP];   // 34 KB
  __shared__ __align__(16) unsigned short Ps[8][16][VP];    // 34 KB

  const int tid = threadIdx.x, wv = tid >> 5, lane = tid & 31;
  const int hi = lane >> 4, l16 = lane & 15;
  const int bh = blockIdx.y;
  const int b = bh >> 4, h = bh & 15;
  const int qb = blockIdx.x * 128;
  const int qw = qb + wv * 16;                   // this wave's first query row
  const size_t base = (size_t)bh * T * D;        // same size for K and V^T

  // Issue K (128 rows x 64) + V^T (64 rows x 128) tile copy.
  auto copy_tile = [&](int buf, int kb) {
#pragma unroll
    for (int it = 0; it < 4; ++it) {
      const int idx = it * 256 + tid;            // 1024 x 16B chunks per buffer
      const int krow = idx >> 3, kcol = (idx & 7) << 3;
      const int vrow = idx >> 4, vcol = (idx & 15) << 3;
      const unsigned short* gk = Kg + base + (size_t)(kb + krow) * D + kcol;
      const unsigned short* gv = Vt + base + (size_t)vrow * T + (kb + vcol);
#if HAVE_ASYNC_LDS
      __builtin_amdgcn_global_load_async_to_lds_b128(
          (AS_GLOBAL v4i*)(AS_GLOBAL void*)(void*)gk,
          (AS_LOCAL v4i*)(AS_LOCAL void*)(void*)&Ks[buf][krow][kcol], 0, 0);
      __builtin_amdgcn_global_load_async_to_lds_b128(
          (AS_GLOBAL v4i*)(AS_GLOBAL void*)(void*)gv,
          (AS_LOCAL v4i*)(AS_LOCAL void*)(void*)&VsT[buf][vrow][vcol], 0, 0);
#else
      *(ushort4*)&Ks[buf][krow][kcol + 0]  = *(const ushort4*)(gk + 0);
      *(ushort4*)&Ks[buf][krow][kcol + 4]  = *(const ushort4*)(gk + 4);
      *(ushort4*)&VsT[buf][vrow][vcol + 0] = *(const ushort4*)(gv + 0);
      *(ushort4*)&VsT[buf][vrow][vcol + 4] = *(const ushort4*)(gv + 4);
#endif
    }
  };

  // Q A-fragments (16 rows x 64 d), resident for the whole kernel.
  v16bf aq[2];
  {
    const unsigned short* qp = Q + base + (size_t)(qw + l16) * D;
#pragma unroll
    for (int c = 0; c < 2; ++c)
#pragma unroll
      for (int e = 0; e < 16; ++e) {
        const int dl = e + hi * 16;
        aq[c][e] = bf16_from_raw(qp[c * 32 + dl]);
      }
  }

  v8f o[4] = {};
  float mrun[8], lrun[8];
#pragma unroll
  for (int r = 0; r < 8; ++r) { mrun[r] = -INFINITY; lrun[r] = 0.f; }

  const int ntiles = blockIdx.x + 1;             // causal: keys <= qb+127
  copy_tile(0, 0);
  for (int j = 0; j < ntiles; ++j) {
    const int cur = j & 1;
    async_wait_all();                            // this wave's DMA done
    __syncthreads();                             // all waves' tiles visible
    if (j + 1 < ntiles) copy_tile(cur ^ 1, (j + 1) * 128);

    const int kb = j * 128;

    // S = Q * K^T : 8 n-subtiles of 16 keys, contraction over d in 2 chunks.
    v8f s[8];
#pragma unroll
    for (int nt = 0; nt < 8; ++nt) {
      v8f accs = {};
#pragma unroll
      for (int c = 0; c < 2; ++c) {
        v16bf bk;
#pragma unroll
        for (int e = 0; e < 16; ++e)
          bk[e] = bf16_from_raw(Ks[cur][nt * 16 + l16][c * 32 + e + hi * 16]);
        accs = wmma_bf16(aq[c], bk, accs);
      }
      s[nt] = accs;
    }

    // Causal mask: only the diagonal tile needs it (uniform branch).
    if (j + 1 == ntiles) {
#pragma unroll
      for (int nt = 0; nt < 8; ++nt) {
        const int kg = kb + nt * 16 + l16;
#pragma unroll
        for (int r = 0; r < 8; ++r)
          if (kg > qw + hi * 8 + r) s[nt][r] = -INFINITY;
      }
    }

    // Online softmax. Row (r + 8*hi) lives across the 16 lanes of one half.
#pragma unroll
    for (int r = 0; r < 8; ++r) {
      float mx = s[0][r];
#pragma unroll
      for (int nt = 1; nt < 8; ++nt) mx = fmaxf(mx, s[nt][r]);
      mx = fmaxf(mx, __shfl_xor(mx, 1, 32));
      mx = fmaxf(mx, __shfl_xor(mx, 2, 32));
      mx = fmaxf(mx, __shfl_xor(mx, 4, 32));
      mx = fmaxf(mx, __shfl_xor(mx, 8, 32));
      const float mn = fmaxf(mrun[r], mx);
      float sum = 0.f;
#pragma unroll
      for (int nt = 0; nt < 8; ++nt) {
        const float p = __expf(s[nt][r] - mn);
        s[nt][r] = p;
        sum += p;
      }
      sum += __shfl_xor(sum, 1, 32);
      sum += __shfl_xor(sum, 2, 32);
      sum += __shfl_xor(sum, 4, 32);
      sum += __shfl_xor(sum, 8, 32);
      const float alpha = __expf(mrun[r] - mn);
      lrun[r] = lrun[r] * alpha + sum;
      mrun[r] = mn;
#pragma unroll
      for (int dt = 0; dt < 4; ++dt) o[dt][r] *= alpha;
    }

    // P: C/D layout -> A layout via per-wave LDS round trip.
#pragma unroll
    for (int nt = 0; nt < 8; ++nt)
#pragma unroll
      for (int r = 0; r < 8; ++r)
        Ps[wv][r + 8 * hi][nt * 16 + l16] = f32_to_bf16_raw(s[nt][r]);
    __syncthreads();

    // O += P (16x128) * V (128x64); V^T rows are key-contiguous.
#pragma unroll
    for (int c = 0; c < 4; ++c) {
      v16bf ap;
#pragma unroll
      for (int e = 0; e < 16; ++e) {
        const int kl = e + hi * 16;
        ap[e] = bf16_from_raw(Ps[wv][l16][c * 32 + kl]);
      }
#pragma unroll
      for (int dt = 0; dt < 4; ++dt) {
        v16bf bvv;
#pragma unroll
        for (int e = 0; e < 16; ++e)
          bvv[e] = bf16_from_raw(VsT[cur][dt * 16 + l16][c * 32 + e + hi * 16]);
        o[dt] = wmma_bf16(ap, bvv, o[dt]);
      }
    }
  }

  // O /= l ; write bf16 attn output in [B, T, C] with column h*64+d.
#pragma unroll
  for (int dt = 0; dt < 4; ++dt) {
    const int d = dt * 16 + l16;
#pragma unroll
    for (int r = 0; r < 8; ++r) {
      const int t = qw + r + 8 * hi;
      const float val = o[dt][r] / lrun[r];
      Og[((size_t)b * T + t) * C + h * D + d] = f32_to_bf16_raw(val);
    }
  }
}

// ---------------------------------------------------------------------------
// Harness entry: x, W_qkv, b_qkv, W_out, b_out (all fp32). Output fp32 [B,T,C].
// Workspace: q/k bf16 [B,H,T,D], v bf16 [B,H,D,T] (16 MB each) +
//            attn bf16 [B,T,C] (16 MB).
// ---------------------------------------------------------------------------
extern "C" void kernel_launch(void* const* d_in, const int* in_sizes, int n_in,
                              void* d_out, int out_size, void* d_ws, size_t ws_size,
                              hipStream_t stream) {
  (void)in_sizes; (void)n_in; (void)out_size; (void)ws_size;
  const float* x    = (const float*)d_in[0];
  const float* Wqkv = (const float*)d_in[1];
  const float* bqkv = (const float*)d_in[2];
  const float* Wout = (const float*)d_in[3];
  const float* bout = (const float*)d_in[4];
  float* out = (float*)d_out;

  constexpr size_t NE = (size_t)4 * 16 * 2048 * 64;  // 8Mi elements
  unsigned short* q   = (unsigned short*)d_ws;
  unsigned short* k   = q + NE;
  unsigned short* vt  = k + NE;
  unsigned short* att = vt + NE;

  const dim3 blk(256);
  // QKV projection: M=8192, N=3072, K=1024
  gemm_wmma_bf16<0><<<dim3(64, 24), blk, 0, stream>>>(
      (const void*)x, Wqkv, bqkv, q, k, vt, nullptr, 8192, 3072, 1024);
  // Causal flash attention: grid (T/128, B*H)
  attn_wmma<<<dim3(16, 64), blk, 0, stream>>>(q, k, vt, att);
  // Output projection: M=8192, N=1024, K=1024
  gemm_wmma_bf16<1><<<dim3(64, 8), blk, 0, stream>>>(
      (const void*)att, Wout, bout, nullptr, nullptr, nullptr, out, 8192, 1024, 1024);
}
